// MultiHeadAttention_37417755082879
// MI455X (gfx1250) — matrix operand order repaired and checked by reading, not compiled, not run on hardware
//
#include <hip/hip_runtime.h>
#include <type_traits>

#define DEV __device__ __forceinline__

typedef __attribute__((ext_vector_type(16))) __bf16       v16bf;
typedef __attribute__((ext_vector_type(8)))  float        v8f;
typedef __attribute__((ext_vector_type(4)))  unsigned int u32x4;
typedef __attribute__((ext_vector_type(4)))  int          v4i;

static constexpr int B_   = 2;
static constexpr int S_   = 2048;
static constexpr int D_   = 768;
static constexpr int H_   = 12;
static constexpr int DK_  = 64;
static constexpr int MTOT = B_ * S_;          // 4096 rows for projection GEMMs
static constexpr float SCL  = 0.125f;         // 1/sqrt(64)
static constexpr float NEGI = -1.0e9f;

// ---- CDNA5 async global->LDS path (guarded; falls back to sync copy) -----
#if __has_builtin(__builtin_amdgcn_global_load_async_to_lds_b128) && \
    __has_builtin(__builtin_amdgcn_s_wait_asynccnt)
#define USE_ASYNC 1
#else
#define USE_ASYNC 0
#endif

DEV void async_cp16(const unsigned short* g, unsigned short* l) {
#if USE_ASYNC
  __builtin_amdgcn_global_load_async_to_lds_b128(
      (__attribute__((address_space(1))) v4i*)g,
      (__attribute__((address_space(3))) v4i*)l, 0, 0);
#else
  *(u32x4*)l = *(const u32x4*)g;
#endif
}
DEV void async_wait_all() {
#if USE_ASYNC
  __builtin_amdgcn_s_wait_asynccnt(0);
#endif
}
// Compiler-level barrier: same-wave LDS ops execute in order at the LDS
// (ISA: DS ops per wave are in order), we only need to stop reordering.
DEV void wave_lds_fence() { asm volatile("" ::: "memory"); }

// ---- helpers -------------------------------------------------------------

DEV unsigned short f2bf(float f) {            // fp32 -> bf16 (RNE)
  union { float f; unsigned u; } v; v.f = f;
  unsigned r = v.u + 0x7FFFu + ((v.u >> 16) & 1u);
  return (unsigned short)(r >> 16);
}

union Frag16 { v16bf v; u32x4 q[2]; unsigned short h[16]; };
union FragC  { v8f   v; float f[8]; };
union Pack8  { u32x4 q; unsigned short h[8]; };

DEV v8f wmma_bf16(const Frag16& a, const Frag16& b, const v8f& c) {
  return __builtin_amdgcn_wmma_f32_16x16x32_bf16(
      false, a.v, false, b.v, (short)0, c, false, false);
}

// ---- fp32 -> bf16 bulk convert (for weight matrices) ---------------------

__global__ __launch_bounds__(256)
void cvt_bf16_kernel(const float* __restrict__ src, unsigned short* __restrict__ dst, int n4)
{
  const int i = blockIdx.x * 256 + threadIdx.x;
  if (i < n4) {
    const float4 f = ((const float4*)src)[i];
    union { unsigned long long u; unsigned short h[4]; } p;
    p.h[0] = f2bf(f.x); p.h[1] = f2bf(f.y); p.h[2] = f2bf(f.z); p.h[3] = f2bf(f.w);
    *(unsigned long long*)(dst + 4 * (size_t)i) = p.u;
  }
}

// ---- GEMM: out[m,n] = sum_k X[m,k]*Wb[n,k] + bias[n] ---------------------
// M = 4096, N = K = 768, Wb already bf16. 128 threads (4 waves), 64x64 tile,
// per-wave 32x32 (2x2 WMMA C-frags). Double-buffered LDS tiles, one barrier
// per K-step; B staged with async global->LDS when available.

template <typename TIN, bool OUT_BF16>
__global__ __launch_bounds__(128)
void proj_kernel(const TIN* __restrict__ X, const unsigned short* __restrict__ Wb,
                 const float* __restrict__ bias, void* __restrict__ outp)
{
  __shared__ unsigned short Atile[2][64 * 40];
  __shared__ unsigned short Btile[2][64 * 40];

  const int m0 = blockIdx.x * 64;
  const int n0 = blockIdx.y * 64;
  const int t    = threadIdx.x;
  const int wave = t >> 5;
  const int lane = t & 31;
  const int wm = (wave >> 1) * 32;
  const int wn = (wave & 1) * 32;
  const int lhalf = lane >> 4, lmod = lane & 15;

  FragC acc[2][2];
  for (int mt = 0; mt < 2; ++mt)
    for (int nt = 0; nt < 2; ++nt)
      for (int r = 0; r < 8; ++r) acc[mt][nt].f[r] = 0.0f;

  const int sr = t >> 1;               // staging row 0..63
  const int sc = (t & 1) * 16;         // staging col 0 or 16

  auto stage = [&](int kt, int buf) {
    const int k0 = kt * 32;
    // A (X): convert fp32 on the fly, or straight copy when already bf16
    if constexpr (std::is_same<TIN, unsigned short>::value) {
      const unsigned short* src = X + (size_t)(m0 + sr) * D_ + k0 + sc;
      async_cp16(src,     &Atile[buf][sr * 40 + sc]);
      async_cp16(src + 8, &Atile[buf][sr * 40 + sc + 8]);
    } else {
      const float* src = X + (size_t)(m0 + sr) * D_ + k0 + sc;
      Pack8 pa0, pa1;
      for (int i = 0; i < 8; ++i) pa0.h[i] = f2bf(src[i]);
      for (int i = 0; i < 8; ++i) pa1.h[i] = f2bf(src[8 + i]);
      *(u32x4*)&Atile[buf][sr * 40 + sc]     = pa0.q;
      *(u32x4*)&Atile[buf][sr * 40 + sc + 8] = pa1.q;
    }
    // B (Wb, bf16): straight copy
    const unsigned short* wsrc = Wb + (size_t)(n0 + sr) * D_ + k0 + sc;
    async_cp16(wsrc,     &Btile[buf][sr * 40 + sc]);
    async_cp16(wsrc + 8, &Btile[buf][sr * 40 + sc + 8]);
  };

  stage(0, 0);

  for (int kt = 0; kt < D_ / 32; ++kt) {
    async_wait_all();
    __syncthreads();
    const int cur = kt & 1;
    if (kt + 1 < D_ / 32) stage(kt + 1, cur ^ 1);

    Frag16 a[2], b[2];
    const int kb = lhalf * 8;
    for (int mt = 0; mt < 2; ++mt) {
      const int row = wm + mt * 16 + lmod;
      a[mt].q[0] = *(const u32x4*)&Atile[cur][row * 40 + kb];
      a[mt].q[1] = *(const u32x4*)&Atile[cur][row * 40 + kb + 16];
    }
    const int cb = lhalf * 8;
    for (int nt = 0; nt < 2; ++nt) {
      const int row = wn + nt * 16 + lmod;
      b[nt].q[0] = *(const u32x4*)&Btile[cur][row * 40 + cb];
      b[nt].q[1] = *(const u32x4*)&Btile[cur][row * 40 + cb + 16];
    }
    for (int mt = 0; mt < 2; ++mt)
      for (int nt = 0; nt < 2; ++nt)
        acc[mt][nt].v = wmma_bf16(a[mt], b[nt], acc[mt][nt].v);
  }

  // epilogue: bias + store
  for (int nt = 0; nt < 2; ++nt) {
    const int gn = n0 + wn + nt * 16 + lmod;
    const float bv = bias[gn];
    for (int mt = 0; mt < 2; ++mt)
      for (int r = 0; r < 8; ++r) {
        const int gm = m0 + wm + mt * 16 + r + lhalf * 8;
        const float val = acc[mt][nt].f[r] + bv;
        if constexpr (OUT_BF16)
          ((unsigned short*)outp)[(size_t)gm * D_ + gn] = f2bf(val);
        else
          ((float*)outp)[(size_t)gm * D_ + gn] = val;
      }
  }
}

// ---- Flash attention -----------------------------------------------------
// Grid (S/64, H, B), 128 threads. Each wave: 16 query rows, online softmax,
// 32 key rows per iteration, double-buffered K/V tiles, 1 barrier/iter.

__global__ __launch_bounds__(128)
void flash_kernel(const unsigned short* __restrict__ q,
                  const unsigned short* __restrict__ k,
                  const unsigned short* __restrict__ v,
                  const int* __restrict__ mask,
                  unsigned short* __restrict__ ctx)
{
  __shared__ unsigned short Klds[2][32 * 72];     // [s_local][d], padded
  __shared__ unsigned short Vlds[2][64 * 40];     // transposed: [d][s_local]
  __shared__ unsigned short Plds[4 * 16 * 40];    // per-wave P tile (bf16)

  const int qt = blockIdx.x, h = blockIdx.y, b = blockIdx.z;
  const int t    = threadIdx.x;
  const int wave = t >> 5;
  const int lane = t & 31;
  const int lhalf = lane >> 4, lmod = lane & 15;
  const int q0 = qt * 64 + wave * 16;             // seq row within batch

  // Q fragments for this wave's 16 rows (held for the whole pass)
  Frag16 aq[2];
  {
    const unsigned short* qp = q + (size_t)(b * S_ + q0 + lmod) * D_ + h * DK_;
    const int kb = lhalf * 8;
    for (int j = 0; j < 2; ++j) {
      aq[j].q[0] = *(const u32x4*)(qp + j * 32 + kb);
      aq[j].q[1] = *(const u32x4*)(qp + j * 32 + kb + 16);
    }
  }

  FragC O[4];
  float mi[8], li[8];
  for (int nf = 0; nf < 4; ++nf)
    for (int r = 0; r < 8; ++r) O[nf].f[r] = 0.0f;
  for (int r = 0; r < 8; ++r) { mi[r] = -INFINITY; li[r] = 0.0f; }

  const int* mbase = mask + (size_t)b * S_ * S_;
  const int sr = t >> 2;               // staging row 0..31
  const int sc = (t & 3) * 16;         // staging col 0/16/32/48

  auto stage = [&](int kt, int buf) {
    const int s0 = kt * 32;
    const size_t gro = (size_t)(b * S_ + s0 + sr) * D_ + h * DK_ + sc;
    // K: straight 16B copies (async-eligible)
    async_cp16(k + gro,     &Klds[buf][sr * 72 + sc]);
    async_cp16(k + gro + 8, &Klds[buf][sr * 72 + sc + 8]);
    // V: transpose while staging
    Pack8 v0, v1;
    v0.q = *(const u32x4*)(v + gro);
    v1.q = *(const u32x4*)(v + gro + 8);
    for (int i = 0; i < 8; ++i) Vlds[buf][(sc + i)     * 40 + sr] = v0.h[i];
    for (int i = 0; i < 8; ++i) Vlds[buf][(sc + 8 + i) * 40 + sr] = v1.h[i];
  };

  stage(0, 0);

  for (int kt = 0; kt < S_ / 32; ++kt) {
    async_wait_all();
    __syncthreads();
    const int cur = kt & 1;
    const int s0 = kt * 32;
    if (kt + 1 < S_ / 32) stage(kt + 1, cur ^ 1);

    // ---- scores: 16 (q) x 32 (k) over d=64 ----
    FragC s0f, s1f;
    for (int r = 0; r < 8; ++r) { s0f.f[r] = 0.0f; s1f.f[r] = 0.0f; }
    for (int j = 0; j < 2; ++j) {
      Frag16 bk0, bk1;
      const int cbk = j * 32 + lhalf * 8;
      bk0.q[0] = *(const u32x4*)&Klds[cur][lmod * 72 + cbk];
      bk0.q[1] = *(const u32x4*)&Klds[cur][lmod * 72 + cbk + 16];
      bk1.q[0] = *(const u32x4*)&Klds[cur][(16 + lmod) * 72 + cbk];
      bk1.q[1] = *(const u32x4*)&Klds[cur][(16 + lmod) * 72 + cbk + 16];
      s0f.v = wmma_bf16(aq[j], bk0, s0f.v);
      s1f.v = wmma_bf16(aq[j], bk1, s1f.v);
    }

    // ---- mask + online softmax, write P (bf16) to this wave's LDS tile ----
    unsigned short* pw = &Plds[wave * 16 * 40];
    for (int r = 0; r < 8; ++r) {
      const int rr   = r + lhalf * 8;
      const int grow = q0 + rr;
      const int* mp  = mbase + (size_t)grow * S_ + s0;
      const int mk0 = mp[lmod], mk1 = mp[16 + lmod];
      float x0 = mk0 ? s0f.f[r] * SCL : NEGI;
      float x1 = mk1 ? s1f.f[r] * SCL : NEGI;
      float loc = fmaxf(x0, x1);
      for (int off = 8; off >= 1; off >>= 1)
        loc = fmaxf(loc, __shfl_xor(loc, off, 32));    // stays in 16-lane half
      const float nm    = fmaxf(mi[r], loc);
      const float alpha = __expf(mi[r] - nm);
      const float p0 = __expf(x0 - nm);
      const float p1 = __expf(x1 - nm);
      float rs = p0 + p1;
      for (int off = 8; off >= 1; off >>= 1)
        rs += __shfl_xor(rs, off, 32);
      li[r] = li[r] * alpha + rs;
      mi[r] = nm;
      for (int nf = 0; nf < 4; ++nf) O[nf].f[r] *= alpha;
      pw[rr * 40 + lmod]      = f2bf(p0);
      pw[rr * 40 + 16 + lmod] = f2bf(p1);
    }
    // Same-wave producer/consumer through LDS: per-wave DS ops are in order,
    // only stop the compiler from reordering. (No block barrier needed.)
    wave_lds_fence();

    // ---- P (16x32) x V (32x64) accumulated into O ----
    Frag16 ap;
    {
      const int kb = lhalf * 8;
      ap.q[0] = *(const u32x4*)&pw[lmod * 40 + kb];
      ap.q[1] = *(const u32x4*)&pw[lmod * 40 + kb + 16];
    }
    for (int nt = 0; nt < 4; ++nt) {
      Frag16 bv;
      const int row = nt * 16 + lmod;
      const int cb  = lhalf * 8;
      bv.q[0] = *(const u32x4*)&Vlds[cur][row * 40 + cb];
      bv.q[1] = *(const u32x4*)&Vlds[cur][row * 40 + cb + 16];
      O[nt].v = wmma_bf16(ap, bv, O[nt].v);
    }
  }

  // ---- normalize and store context (bf16, layout [B*S, H*DK]) ----
  for (int r = 0; r < 8; ++r) {
    const int rr   = r + lhalf * 8;
    const int grow = q0 + rr;
    const float inv = 1.0f / li[r];
    for (int nt = 0; nt < 4; ++nt) {
      const int d = nt * 16 + lmod;
      ctx[(size_t)(b * S_ + grow) * D_ + h * DK_ + d] = f2bf(O[nt].f[r] * inv);
    }
  }
}

// ---- launcher ------------------------------------------------------------

extern "C" void kernel_launch(void* const* d_in, const int* in_sizes, int n_in,
                              void* d_out, int out_size, void* d_ws, size_t ws_size,
                              hipStream_t stream)
{
  const float* query = (const float*)d_in[0];
  const float* key   = (const float*)d_in[1];
  const float* value = (const float*)d_in[2];
  const int*   mask  = (const int*)  d_in[3];
  const float* Wq = (const float*)d_in[4];
  const float* bq = (const float*)d_in[5];
  const float* Wk = (const float*)d_in[6];
  const float* bk = (const float*)d_in[7];
  const float* Wv = (const float*)d_in[8];
  const float* bv = (const float*)d_in[9];
  const float* Wo = (const float*)d_in[10];
  const float* bo = (const float*)d_in[11];

  // workspace layout (bf16): q,k,v,ctx activations + 4 converted weights
  unsigned short* qb  = (unsigned short*)d_ws;
  unsigned short* kb  = qb  + (size_t)MTOT * D_;
  unsigned short* vb  = kb  + (size_t)MTOT * D_;
  unsigned short* cb  = vb  + (size_t)MTOT * D_;
  unsigned short* wqb = cb  + (size_t)MTOT * D_;
  unsigned short* wkb = wqb + (size_t)D_ * D_;
  unsigned short* wvb = wkb + (size_t)D_ * D_;
  unsigned short* wob = wvb + (size_t)D_ * D_;

  const int n4 = D_ * D_ / 4;
  dim3 gc((n4 + 255) / 256), bc(256);
  hipLaunchKernelGGL(cvt_bf16_kernel, gc, bc, 0, stream, Wq, wqb, n4);
  hipLaunchKernelGGL(cvt_bf16_kernel, gc, bc, 0, stream, Wk, wkb, n4);
  hipLaunchKernelGGL(cvt_bf16_kernel, gc, bc, 0, stream, Wv, wvb, n4);
  hipLaunchKernelGGL(cvt_bf16_kernel, gc, bc, 0, stream, Wo, wob, n4);

  dim3 gp(MTOT / 64, D_ / 64), bp(128);
  hipLaunchKernelGGL((proj_kernel<float, true>), gp, bp, 0, stream, query, wqb, bq, (void*)qb);
  hipLaunchKernelGGL((proj_kernel<float, true>), gp, bp, 0, stream, key,   wkb, bk, (void*)kb);
  hipLaunchKernelGGL((proj_kernel<float, true>), gp, bp, 0, stream, value, wvb, bv, (void*)vb);

  dim3 gf(S_ / 64, H_, B_), bfk(128);
  hipLaunchKernelGGL(flash_kernel, gf, bfk, 0, stream, qb, kb, vb, mask, cb);

  hipLaunchKernelGGL((proj_kernel<unsigned short, false>), gp, bp, 0, stream, cb, wob, bo, d_out);
}
